// STGAMAM_31267361915432
// MI455X (gfx1250) — matrix operand order repaired and checked by reading, not compiled, not run on hardware
//
#include <hip/hip_runtime.h>
#include <hip/hip_bf16.h>
#include <stdint.h>

// ---------------------------------------------------------------------------
// MI455X (gfx1250) implementation.
//
// Heavy GEMMs (QKV / out-proj / FF1 / FF2, ~38.6 GFLOP, ~200MB fp32 weights)
// run on v_wmma_f32_16x16x32_f16 with f32 accumulation:
//   - weights converted fp32 -> f16, row-major (M x Kp), zero-padded
//   - activations transposed+converted to f16 feature-major (Kp x 384)
//   A fragment: lane -> M row, vgpr pair -> K  (ISA 16-bit A 16x32 table)
//   B fragment: lane&15 -> N col, vgpr pair -> K  (ISA B-matrix tables)
//   C/D: vgpr p -> M = p + 8*(lane>>4), lane&15 -> N
// Each wave holds NB=3 N-tile accumulators so every A (weight) fragment is
// reused 3x; A/B pointers are strength-reduced so the K-loop is pure
// load+wmma (no 64-bit multiplies in the hot loop).
// Attention / layernorm / GAT / decoder are small and stay in fp32 VALU.
// The per-timestep GAT in the reference is dead code (output unused) -> skipped.
// ---------------------------------------------------------------------------

typedef __attribute__((ext_vector_type(16))) _Float16 v16h;
typedef __attribute__((ext_vector_type(8)))  float    v8f;

#define TT 48
#define NNODE 8
#define RR 384          // T*N rows
#define DD 2044
#define TD 6132         // 3*D
#define HH 7
#define HD 292          // D/H
#define FF 2048
#define KP 2048         // padded K for all GEMMs
#define NB 3            // N-tiles per wave (24 % 3 == 0)

// ---- fp32 -> f16 weight conversion with zero padding (Mp x Kp) ----
__global__ void convert_w_kernel(const float* __restrict__ W, _Float16* __restrict__ Wh,
                                 int M, int K, int Mp, int Kp) {
    size_t idx = (size_t)blockIdx.x * blockDim.x + threadIdx.x;
    size_t total = (size_t)Mp * Kp;
    if (idx >= total) return;
    int m = (int)(idx / Kp), k = (int)(idx % Kp);
    float v = (m < M && k < K) ? W[(size_t)m * K + k] : 0.0f;
    Wh[idx] = (_Float16)v;
}

// ---- activations: fp32 row-major (R x ldx) -> f16 feature-major (Kp x R) ----
__global__ void transpose_act_kernel(const float* __restrict__ X, _Float16* __restrict__ XT,
                                     int R, int K, int ldx, int Kp) {
    size_t idx = (size_t)blockIdx.x * blockDim.x + threadIdx.x;
    size_t total = (size_t)Kp * R;
    if (idx >= total) return;
    int k = (int)(idx / R), r = (int)(idx % R);
    XT[idx] = (k < K) ? (_Float16)X[(size_t)r * ldx + k] : (_Float16)0.0f;
}

// ---- WMMA GEMM: out[r][f] = sum_k W[f][k] * X[r][k] (+bias, opt relu) ----
// One wave: 16 M-rows x (3 x 16) N-cols; A fragment loaded once per K-step.
__global__ void gemm_f16_wmma(const _Float16* __restrict__ A,   // Mp x Kp (weights)
                              const _Float16* __restrict__ B,   // Kp x R  (act^T)
                              const float* __restrict__ bias,
                              float* __restrict__ out,          // R x ldo
                              int Mp, int Kp, int R, int ldo, int Mreal, int relu) {
    int wave = threadIdx.x >> 5;
    int lane = threadIdx.x & 31;
    int tilesM = Mp >> 4;
    int ngroups = (R >> 4) / NB;
    int tile = blockIdx.x * (blockDim.x >> 5) + wave;
    if (tile >= tilesM * ngroups) return;         // wave-uniform exit (EXEC stays full)
    int mo = tile % tilesM;
    int ng = tile / tilesM;
    int m = lane & 15;
    int g = lane >> 4;

    const _Float16* aptr = A + (size_t)(mo * 16 + m) * Kp;        // += 32 per K-step
    const _Float16* bptr = B + (size_t)m + ng * (16 * NB);        // += 32*R per K-step
    const size_t bstep = (size_t)32 * R;

    v8f acc0 = {}, acc1 = {}, acc2 = {};
    for (int kb = 0; kb < Kp; kb += 32) {
        union { v16h v; unsigned u[8]; } af, bf0, bf1, bf2;
#pragma unroll
        for (int v = 0; v < 8; ++v) {
            int koff = ((v & 3) << 1) + (g << 3) + ((v >> 2) << 4);
            af.u[v] = *(const unsigned*)(aptr + koff);
        }
#pragma unroll
        for (int v = 0; v < 8; ++v) {
            int koff = ((v & 3) << 1) + (g << 3) + ((v >> 2) << 4);
            const _Float16* bp = bptr + (size_t)koff * R;
            bf0.v[2 * v] = bp[0];      bf0.v[2 * v + 1] = bp[R];
            bf1.v[2 * v] = bp[16];     bf1.v[2 * v + 1] = bp[R + 16];
            bf2.v[2 * v] = bp[32];     bf2.v[2 * v + 1] = bp[R + 32];
        }
        acc0 = __builtin_amdgcn_wmma_f32_16x16x32_f16(
            false, af.v, false, bf0.v, (short)0, acc0, false, false);
        acc1 = __builtin_amdgcn_wmma_f32_16x16x32_f16(
            false, af.v, false, bf1.v, (short)0, acc1, false, false);
        acc2 = __builtin_amdgcn_wmma_f32_16x16x32_f16(
            false, af.v, false, bf2.v, (short)0, acc2, false, false);
        aptr += 32;
        bptr += bstep;
    }
    int rbase = ng * (16 * NB) + m;
#pragma unroll
    for (int p = 0; p < 8; ++p) {
        int f = mo * 16 + p + (g << 3);
        if (f < Mreal) {
            float bv = bias[f];
            float v0 = acc0[p] + bv;
            float v1 = acc1[p] + bv;
            float v2 = acc2[p] + bv;
            if (relu) { v0 = fmaxf(v0, 0.0f); v1 = fmaxf(v1, 0.0f); v2 = fmaxf(v2, 0.0f); }
            out[(size_t)rbase * ldo + f]        = v0;
            out[(size_t)(rbase + 16) * ldo + f] = v1;
            out[(size_t)(rbase + 32) * ldo + f] = v2;
        }
    }
}

// ---- causal multi-head attention; qkv (384 x 6132) -> o (384 x 2044) ----
__global__ void attention_kernel(const float* __restrict__ qkv, float* __restrict__ o) {
    int t = blockIdx.x, h = blockIdx.y, n = blockIdx.z;
    int tid = threadIdx.x;                        // 64 threads
    __shared__ float att[TT];
    int rt = t * NNODE + n;
    const float* q = qkv + (size_t)rt * TD + h * HD;
    const float scale = 1.0f / sqrtf(292.0f);
    if (tid <= t) {
        const float* kk = qkv + (size_t)(tid * NNODE + n) * TD + DD + h * HD;
        float s = 0.0f;
        for (int d = 0; d < HD; ++d) s += q[d] * kk[d];
        att[tid] = s * scale;
    }
    __syncthreads();
    if (tid == 0) {
        float m = -3.4e38f;
        for (int s = 0; s <= t; ++s) m = fmaxf(m, att[s]);
        float sum = 0.0f;
        for (int s = 0; s <= t; ++s) { float e = expf(att[s] - m); att[s] = e; sum += e; }
        float inv = 1.0f / sum;
        for (int s = 0; s <= t; ++s) att[s] *= inv;
    }
    __syncthreads();
    for (int d = tid; d < HD; d += 64) {
        float acc = 0.0f;
        for (int s = 0; s <= t; ++s)
            acc += att[s] * qkv[(size_t)(s * NNODE + n) * TD + 2 * DD + h * HD + d];
        o[(size_t)rt * DD + h * HD + d] = acc;
    }
}

// ---- out = LayerNorm(x + y) * g + b, one block per row ----
__global__ void add_layernorm(const float* __restrict__ x, const float* __restrict__ y,
                              const float* __restrict__ g, const float* __restrict__ b,
                              float* __restrict__ out, int D) {
    int r = blockIdx.x, tid = threadIdx.x;
    __shared__ float red[256];
    const float* xr = x + (size_t)r * D;
    const float* yr = y + (size_t)r * D;
    float s = 0.0f;
    for (int i = tid; i < D; i += 256) s += xr[i] + yr[i];
    red[tid] = s; __syncthreads();
    for (int st = 128; st > 0; st >>= 1) { if (tid < st) red[tid] += red[tid + st]; __syncthreads(); }
    float mean = red[0] / D; __syncthreads();
    float v = 0.0f;
    for (int i = tid; i < D; i += 256) { float z = xr[i] + yr[i] - mean; v += z * z; }
    red[tid] = v; __syncthreads();
    for (int st = 128; st > 0; st >>= 1) { if (tid < st) red[tid] += red[tid + st]; __syncthreads(); }
    float inv = rsqrtf(red[0] / D + 1e-5f); __syncthreads();
    for (int i = tid; i < D; i += 256)
        out[(size_t)r * D + i] = (xr[i] + yr[i] - mean) * inv * g[i] + b[i];
}

// ---- fusion-GAT head (24 nodes, hidden 16), 6 blocks of 384 threads ----
__global__ void gat_head_kernel(const float* __restrict__ X2, const int* __restrict__ adj,
                                const float* __restrict__ W6, const float* __restrict__ a6,
                                float* __restrict__ hcat) {
    const int NN = 24, HID = 16;
    int h = blockIdx.x, tid = threadIdx.x;
    int i = tid >> 4, j = tid & 15;
    const float* W = W6 + (size_t)h * DD * HID;
    const float* a = a6 + h * 32;
    __shared__ float wh[24][16];
    __shared__ float w1s[24], w2s[24];
    __shared__ float att[24][24];
    float s = 0.0f;
    const float* xi = X2 + (size_t)i * DD;
    for (int f = 0; f < DD; ++f) s += xi[f] * W[(size_t)f * HID + j];
    wh[i][j] = s;
    __syncthreads();
    if (tid < NN) {
        float s1 = 0.0f, s2 = 0.0f;
        for (int c = 0; c < HID; ++c) { s1 += wh[tid][c] * a[c]; s2 += wh[tid][c] * a[HID + c]; }
        w1s[tid] = s1; w2s[tid] = s2;
    }
    __syncthreads();
    for (int idx = tid; idx < NN * NN; idx += blockDim.x) {
        int ii = idx / NN, jj = idx % NN;
        float e = w1s[ii] + w2s[jj];
        e = e > 0.0f ? e : 0.2f * e;
        att[ii][jj] = (adj[ii * NN + jj] > 0) ? e : -9e15f;
    }
    __syncthreads();
    if (tid < NN) {
        float m = -3.4e38f;
        for (int jj = 0; jj < NN; ++jj) m = fmaxf(m, att[tid][jj]);
        float sum = 0.0f;
        for (int jj = 0; jj < NN; ++jj) { float e = expf(att[tid][jj] - m); att[tid][jj] = e; sum += e; }
        float inv = 1.0f / sum;
        for (int jj = 0; jj < NN; ++jj) att[tid][jj] *= inv;
    }
    __syncthreads();
    float acc = 0.0f;
    for (int jj = 0; jj < NN; ++jj) acc += att[i][jj] * wh[jj][j];
    float e = acc > 0.0f ? acc : (expf(acc) - 1.0f);          // elu (concat head)
    hcat[(size_t)i * 96 + h * HID + j] = e;
}

// ---- WhO = hcat(24x96) @ Wout(96x2044) ----
__global__ void gat_out_mm(const float* __restrict__ hcat, const float* __restrict__ Wout,
                           float* __restrict__ who) {
    int i = blockIdx.x;
    int f = blockIdx.y * 256 + threadIdx.x;
    if (f >= DD) return;
    float s = 0.0f;
    for (int c = 0; c < 96; ++c) s += hcat[i * 96 + c] * Wout[(size_t)c * DD + f];
    who[(size_t)i * DD + f] = s;
}

__global__ void gat_aout_kernel(const float* __restrict__ who, const float* __restrict__ aout,
                                float* __restrict__ w1o, float* __restrict__ w2o) {
    int i = blockIdx.x, tid = threadIdx.x;
    __shared__ float r1[256], r2[256];
    float s1 = 0.0f, s2 = 0.0f;
    for (int f = tid; f < DD; f += 256) {
        float w = who[(size_t)i * DD + f];
        s1 += w * aout[f]; s2 += w * aout[DD + f];
    }
    r1[tid] = s1; r2[tid] = s2; __syncthreads();
    for (int st = 128; st > 0; st >>= 1) {
        if (tid < st) { r1[tid] += r1[tid + st]; r2[tid] += r2[tid + st]; }
        __syncthreads();
    }
    if (tid == 0) { w1o[i] = r1[0]; w2o[i] = r2[0]; }
}

__global__ void gat_att2_kernel(const float* __restrict__ w1o, const float* __restrict__ w2o,
                                const int* __restrict__ adj, float* __restrict__ att) {
    const int NN = 24;
    __shared__ float e[24][24];
    int tid = threadIdx.x;                       // 576 threads
    if (tid < NN * NN) {
        int i = tid / NN, j = tid % NN;
        float v = w1o[i] + w2o[j];
        v = v > 0.0f ? v : 0.2f * v;
        e[i][j] = (adj[i * NN + j] > 0) ? v : -9e15f;
    }
    __syncthreads();
    if (tid < NN) {
        float m = -3.4e38f;
        for (int j = 0; j < NN; ++j) m = fmaxf(m, e[tid][j]);
        float s = 0.0f;
        for (int j = 0; j < NN; ++j) { float ex = expf(e[tid][j] - m); e[tid][j] = ex; s += ex; }
        float inv = 1.0f / s;
        for (int j = 0; j < NN; ++j) att[tid * NN + j] = e[tid][j] * inv;
    }
}

__global__ void gat_hp_kernel(const float* __restrict__ att, const float* __restrict__ who,
                              float* __restrict__ y) {
    int i = blockIdx.x;
    int f = blockIdx.y * 256 + threadIdx.x;
    if (f >= DD) return;
    float s = 0.0f;
    for (int j = 0; j < 24; ++j) s += att[i * 24 + j] * who[(size_t)j * DD + f];
    y[(size_t)i * DD + f] = s > 0.0f ? s : (expf(s) - 1.0f);  // elu (no concat)
}

// ---- log_softmax of rows 16..23 fused with the decoder dot product ----
__global__ void dec_kernel(const float* __restrict__ y, const float* __restrict__ dw,
                           const float* __restrict__ db, float* __restrict__ out) {
    int n = blockIdx.x, tid = threadIdx.x;
    const float* yr = y + (size_t)(16 + n) * DD;
    __shared__ float red[256];
    float m = -3.4e38f;
    for (int f = tid; f < DD; f += 256) m = fmaxf(m, yr[f]);
    red[tid] = m; __syncthreads();
    for (int st = 128; st > 0; st >>= 1) { if (tid < st) red[tid] = fmaxf(red[tid], red[tid + st]); __syncthreads(); }
    m = red[0]; __syncthreads();
    float s = 0.0f;
    for (int f = tid; f < DD; f += 256) s += expf(yr[f] - m);
    red[tid] = s; __syncthreads();
    for (int st = 128; st > 0; st >>= 1) { if (tid < st) red[tid] += red[tid + st]; __syncthreads(); }
    float lse = logf(red[0]); __syncthreads();
    float d = 0.0f;
    for (int f = tid; f < DD; f += 256) d += (yr[f] - m - lse) * dw[f];
    red[tid] = d; __syncthreads();
    for (int st = 128; st > 0; st >>= 1) { if (tid < st) red[tid] += red[tid + st]; __syncthreads(); }
    if (tid == 0) out[n] = red[0] + db[0];
}

// ---- host-side helper: convert weights + transpose acts + WMMA GEMM ----
static void run_gemm(const float* W, const float* bias, const float* Xin,
                     int M, int K, int Mp,
                     float* out, int ldo, int relu,
                     _Float16* WH, _Float16* XT, int ldx, hipStream_t stream) {
    size_t nw = (size_t)Mp * KP;
    convert_w_kernel<<<(unsigned)((nw + 255) / 256), 256, 0, stream>>>(W, WH, M, K, Mp, KP);
    size_t nt = (size_t)KP * RR;
    transpose_act_kernel<<<(unsigned)((nt + 255) / 256), 256, 0, stream>>>(Xin, XT, RR, K, ldx, KP);
    int tiles = (Mp / 16) * ((RR / 16) / NB);
    gemm_f16_wmma<<<(tiles + 7) / 8, 256, 0, stream>>>(WH, XT, bias, out, Mp, KP, RR, ldo, M, relu);
}

extern "C" void kernel_launch(void* const* d_in, const int* in_sizes, int n_in,
                              void* d_out, int out_size, void* d_ws, size_t ws_size,
                              hipStream_t stream) {
    const float* src    = (const float*)d_in[0];
    const int*   st_adj = (const int*)d_in[2];   // d_in[1] (adj) feeds only dead code

    char* ws = (char*)d_ws;
    size_t off = 0;
    auto carve = [&](size_t bytes) -> void* {
        void* p = (void*)(ws + off);
        off += (bytes + 255) & ~(size_t)255;
        return p;
    };
    _Float16* WH  = (_Float16*)carve((size_t)6144 * KP * 2); // reused per GEMM
    _Float16* XT  = (_Float16*)carve((size_t)KP * RR * 2);
    float* QKV  = (float*)carve((size_t)RR * TD * 4);
    float* OATT = (float*)carve((size_t)RR * DD * 4);
    float* PROJ = (float*)carve((size_t)RR * DD * 4);
    float* Y    = (float*)carve((size_t)RR * DD * 4);
    float* H1   = (float*)carve((size_t)RR * FF * 4);
    float* H2   = (float*)carve((size_t)RR * DD * 4);
    float* XA   = (float*)carve((size_t)RR * DD * 4);
    float* XB   = (float*)carve((size_t)RR * DD * 4);
    float* HCAT = (float*)carve((size_t)24 * 96 * 4);
    float* WHO  = (float*)carve((size_t)24 * DD * 4);
    float* ATT  = (float*)carve((size_t)24 * 24 * 4);
    float* YG   = (float*)carve((size_t)24 * DD * 4);
    float* W1O  = (float*)carve(24 * 4);
    float* W2O  = (float*)carve(24 * 4);

    const float* Xcur = src;
    float* Xbufs[2] = {XA, XB};
    for (int p = 0; p < 2; ++p) {
        const float* in_w  = (const float*)d_in[3 + p * 12 + 0];
        const float* in_b  = (const float*)d_in[3 + p * 12 + 1];
        const float* out_w = (const float*)d_in[3 + p * 12 + 2];
        const float* out_b = (const float*)d_in[3 + p * 12 + 3];
        const float* w1    = (const float*)d_in[3 + p * 12 + 4];
        const float* b1    = (const float*)d_in[3 + p * 12 + 5];
        const float* w2    = (const float*)d_in[3 + p * 12 + 6];
        const float* b2    = (const float*)d_in[3 + p * 12 + 7];
        const float* g1    = (const float*)d_in[3 + p * 12 + 8];
        const float* be1   = (const float*)d_in[3 + p * 12 + 9];
        const float* g2    = (const float*)d_in[3 + p * 12 + 10];
        const float* be2   = (const float*)d_in[3 + p * 12 + 11];

        run_gemm(in_w, in_b, Xcur, TD, DD, 6144, QKV, TD, 0, WH, XT, DD, stream);
        attention_kernel<<<dim3(TT, HH, NNODE), 64, 0, stream>>>(QKV, OATT);
        run_gemm(out_w, out_b, OATT, DD, DD, 2048, PROJ, DD, 0, WH, XT, DD, stream);
        add_layernorm<<<RR, 256, 0, stream>>>(Xcur, PROJ, g1, be1, Y, DD);
        run_gemm(w1, b1, Y, FF, DD, 2048, H1, FF, 1, WH, XT, DD, stream);
        run_gemm(w2, b2, H1, DD, FF, 2048, H2, DD, 0, WH, XT, FF, stream);
        add_layernorm<<<RR, 256, 0, stream>>>(Y, H2, g2, be2, Xbufs[p], DD);
        Xcur = Xbufs[p];
    }

    const float* W6   = (const float*)d_in[27];
    const float* a6   = (const float*)d_in[28];
    const float* Wout = (const float*)d_in[29];
    const float* aout = (const float*)d_in[30];
    const float* dw   = (const float*)d_in[31];
    const float* db   = (const float*)d_in[32];
    const float* X2 = Xcur + (size_t)360 * DD;   // enc[45:48] -> last 24 rows

    gat_head_kernel<<<6, 384, 0, stream>>>(X2, st_adj, W6, a6, HCAT);
    gat_out_mm<<<dim3(24, 8), 256, 0, stream>>>(HCAT, Wout, WHO);
    gat_aout_kernel<<<24, 256, 0, stream>>>(WHO, aout, W1O, W2O);
    gat_att2_kernel<<<1, 576, 0, stream>>>(W1O, W2O, st_adj, ATT);
    gat_hp_kernel<<<dim3(24, 8), 256, 0, stream>>>(ATT, WHO, YG);
    dec_kernel<<<8, 256, 0, stream>>>(YG, dw, db, (float*)d_out);
}
